// LSTM_81174881894405
// MI455X (gfx1250) — compile-verified, matrix operand order checked by reading, not run on hardware
//
#include <hip/hip_runtime.h>
#include <hip/hip_bf16.h>

// ---------------------------------------------------------------------------
// LSTM for MI455X (gfx1250): bf16 WMMA recurrence, weights resident in LDS,
// persistent grid with global barrier, fused [x_t | h] K=1024 GEMM per step.
// Round 2: x is pre-converted + pre-swizzled into WMMA A-fragment layout by a
// parallel pre-pass (if ws_size allows), removing all f32->bf16 VALU work from
// the sequential 2048-step critical path.
// ---------------------------------------------------------------------------

typedef __attribute__((ext_vector_type(16))) __bf16 v16bf;
typedef __attribute__((ext_vector_type(8)))  float  v8f;

#define B_   64
#define S_   2048
#define I_   512
#define H_   512
#define NCB  32      // workgroups; each owns 16 hidden units (x4 gates = 64 cols)
#define TPB  128     // 4 waves of 32

__device__ __forceinline__ unsigned short f2bf(float f) {
    union { float f; unsigned u; } x; x.f = f;
    unsigned r = x.u + 0x7FFFu + ((x.u >> 16) & 1u);   // round-to-nearest-even
    return (unsigned short)(r >> 16);
}
__device__ __forceinline__ unsigned pk2(float lo, float hi) {
    return (unsigned)f2bf(lo) | ((unsigned)f2bf(hi) << 16);
}
__device__ __forceinline__ unsigned kmap(unsigned v, unsigned half) {
    // 16-bit WMMA operand layout (CDNA5 ISA): VGPR v holds K = kmap, kmap+1
    return 2u * v + (v >= 4u ? 8u : 0u) + 8u * half;
}

// ---------------------------------------------------------------------------
// Pack [Wx;Wh] (1024 x 2048) into bf16, pre-swizzled into the WMMA B-fragment
// per-lane layout: Bsw[cb][ks][gate][lane][v] = pack(W[k][col], W[k+1][col])
// ---------------------------------------------------------------------------
__global__ void lstm_pack(const float* __restrict__ Wxi, const float* __restrict__ Whi,
                          const float* __restrict__ Wxf, const float* __restrict__ Whf,
                          const float* __restrict__ Wxg, const float* __restrict__ Whg,
                          const float* __restrict__ Wxo, const float* __restrict__ Who,
                          unsigned* __restrict__ Bsw) {
    unsigned idx  = blockIdx.x * blockDim.x + threadIdx.x;   // 1,048,576 threads
    unsigned v    = idx & 7u;
    unsigned lane = (idx >> 3) & 31u;
    unsigned gate = (idx >> 8) & 3u;
    unsigned ks   = (idx >> 10) & 31u;
    unsigned cb   = idx >> 15;
    unsigned l    = lane & 15u, half = lane >> 4;
    unsigned jj   = cb * 16u + l;                              // column within gate
    unsigned k    = ks * 32u + kmap(v, half);                  // 0..1022 (even)

    const float* Wx[4] = { Wxi, Wxf, Wxg, Wxo };
    const float* Wh[4] = { Whi, Whf, Whg, Who };
    const float* W = (k < 512u) ? Wx[gate] : Wh[gate];
    unsigned kr = (k < 512u) ? k : (k - 512u);
    float f0 = W[(size_t)kr * 512u + jj];
    float f1 = W[(size_t)(kr + 1u) * 512u + jj];
    Bsw[idx] = pk2(f0, f1);
}

// ---------------------------------------------------------------------------
// Pre-swizzle x into WMMA A-fragment layout (bf16 packed pairs):
//   xsw[row][t][ks*2+half][v] = pack(x[row][t][k], x[row][t][k+1]),
//   k = ks*32 + kmap(v, half).  32M packed u32 = 128 MB.
// Fully parallel, bandwidth-bound; runs off the sequential critical path.
// ---------------------------------------------------------------------------
__global__ void lstm_xpack(const float* __restrict__ x, unsigned* __restrict__ xsw) {
    unsigned idx  = blockIdx.x * blockDim.x + threadIdx.x;     // 33,554,432 threads
    unsigned v    = idx & 7u;
    unsigned sh   = (idx >> 3) & 31u;                          // ks*2 + half
    unsigned t    = (idx >> 8) & 2047u;
    unsigned row  = idx >> 19;
    unsigned ks   = sh >> 1, half = sh & 1u;
    unsigned k    = ks * 32u + kmap(v, half);
    const float* xp = x + (((size_t)row * S_ + t) * I_ + k);
    float2 p = *reinterpret_cast<const float2*>(xp);
    xsw[idx] = pk2(p.x, p.y);
}

// ---------------------------------------------------------------------------
// Combined biases, zero the double-buffered h state, zero the sync counter.
// ---------------------------------------------------------------------------
__global__ void lstm_init(const float* __restrict__ bxi, const float* __restrict__ bhi,
                          const float* __restrict__ bxf, const float* __restrict__ bhf,
                          const float* __restrict__ bxg, const float* __restrict__ bhg,
                          const float* __restrict__ bxo, const float* __restrict__ bho,
                          float* __restrict__ bias, unsigned* __restrict__ hbuf,
                          unsigned* __restrict__ ctr) {
    unsigned idx = blockIdx.x * blockDim.x + threadIdx.x;      // 32768 threads
    if (idx < (unsigned)(2 * B_ * H_ / 2)) hbuf[idx] = 0u;     // both h ping-pong bufs
    if (idx < 4u * H_) {
        unsigned g = idx >> 9, jj = idx & 511u;
        const float* bx = (g == 0) ? bxi : (g == 1) ? bxf : (g == 2) ? bxg : bxo;
        const float* bh = (g == 0) ? bhi : (g == 1) ? bhf : (g == 2) ? bhg : bho;
        bias[idx] = bx[jj] + bh[jj];
    }
    if (idx == 0) *ctr = 0u;
}

// ---------------------------------------------------------------------------
// Persistent recurrence kernel: 32 WGs x 128 threads (4 waves).
//  - WG cb owns hidden units [cb*16, cb*16+16) across all 4 gates -> the c/h
//    update is purely in-register (i,f,g,o accumulators live in one wave).
//  - 128 KB of pre-swizzled weights live in LDS for all 2048 steps.
//  - USE_XSW: A-fragments for x are two global b128 loads (no conversion).
// ---------------------------------------------------------------------------
template <bool USE_XSW>
__launch_bounds__(TPB, 1)
__global__ void lstm_main(const float* __restrict__ x,
                          const unsigned* __restrict__ xsw,
                          const unsigned* __restrict__ Bsw,
                          const float* __restrict__ bias,
                          unsigned* __restrict__ hbuf,       // 2 x [64][256] u32 (bf16)
                          unsigned* __restrict__ ctr,
                          float* __restrict__ out) {
    __shared__ uint4 ldsB[8192];                               // 128 KB weight slice

    const unsigned cb   = blockIdx.x;
    const unsigned tid  = threadIdx.x;
    const unsigned wave = tid >> 5, lane = tid & 31u;
    const unsigned l    = lane & 15u, half = lane >> 4;
    const unsigned m0   = wave * 16u;                          // batch-tile base
    const unsigned hb   = cb * 16u;                            // hidden-slice base

    // Stage this WG's weight fragments into LDS once (coalesced b128 copies).
    const uint4* gB = reinterpret_cast<const uint4*>(Bsw) + (size_t)cb * 8192u;
    for (unsigned i = tid; i < 8192u; i += TPB) ldsB[i] = gB[i];
    __syncthreads();

    float bgate[4];
#pragma unroll
    for (int g = 0; g < 4; ++g) bgate[g] = bias[g * 512 + hb + l];

    float c[8];
#pragma unroll
    for (int v2 = 0; v2 < 8; ++v2) c[v2] = 0.f;

    const unsigned brow = m0 + l;                              // A-fragment batch row
    const float* xrow   = x + (size_t)brow * S_ * I_;
    // xsw as uint4: 64 uint4 per (row,t); lane's slot = (ks*2+half)*2 uint4
    const uint4* xswrow = reinterpret_cast<const uint4*>(xsw)
                        + (size_t)brow * S_ * 64u + half * 2u;
    const unsigned HW   = (unsigned)(B_ * H_ / 2);             // u32 per h buffer

    for (int t = 0; t < S_; ++t) {
        const unsigned* hin = hbuf + (unsigned)(t & 1) * HW + brow * (H_ / 2);
        unsigned short* hout =
            reinterpret_cast<unsigned short*>(hbuf + (unsigned)((t + 1) & 1) * HW);
        const float* xt  = xrow + (size_t)t * I_;
        const uint4* xt4 = xswrow + (size_t)t * 64u;

        v8f acc[4];
#pragma unroll
        for (int g = 0; g < 4; ++g) {
            v8f a;
#pragma unroll
            for (int v2 = 0; v2 < 8; ++v2) a[v2] = bgate[g];   // bias depends on col only
            acc[g] = a;
        }

        // ---- x projection: K = 0..511 ----
        for (int ks = 0; ks < 16; ++ks) {
            union { v16bf v; unsigned u[8]; uint4 q[2]; } afr;
            if (USE_XSW) {
                const uint4* p = xt4 + (unsigned)ks * 4u;      // 2 contiguous b128 loads
                afr.q[0] = p[0]; afr.q[1] = p[1];
            } else {
#pragma unroll
                for (int v2 = 0; v2 < 8; ++v2) {
                    unsigned k = (unsigned)ks * 32u + kmap((unsigned)v2, half);
                    float2 p = *reinterpret_cast<const float2*>(xt + k);
                    afr.u[v2] = pk2(p.x, p.y);
                }
            }
#pragma unroll
            for (int g = 0; g < 4; ++g) {
                union { v16bf v; uint4 q[2]; } bfr;
                const uint4* bp = &ldsB[(((unsigned)ks * 4u + g) * 32u + lane) * 2u];
                bfr.q[0] = bp[0]; bfr.q[1] = bp[1];
                acc[g] = __builtin_amdgcn_wmma_f32_16x16x32_bf16(
                    false, afr.v, false, bfr.v, (short)0, acc[g], false, false);
            }
        }

        // ---- h projection: K = 512..1023 (h already bf16-packed) ----
        for (int ks = 0; ks < 16; ++ks) {
            union { v16bf v; unsigned u[8]; } afr;
#pragma unroll
            for (int v2 = 0; v2 < 8; ++v2) {
                unsigned k = (unsigned)ks * 32u + kmap((unsigned)v2, half);
                afr.u[v2] = hin[k >> 1];
            }
#pragma unroll
            for (int g = 0; g < 4; ++g) {
                union { v16bf v; uint4 q[2]; } bfr;
                const uint4* bp =
                    &ldsB[((((unsigned)ks + 16u) * 4u + g) * 32u + lane) * 2u];
                bfr.q[0] = bp[0]; bfr.q[1] = bp[1];
                acc[g] = __builtin_amdgcn_wmma_f32_16x16x32_bf16(
                    false, afr.v, false, bfr.v, (short)0, acc[g], false, false);
            }
        }

        // ---- gates + state update (f32, all in registers) ----
#pragma unroll
        for (int v2 = 0; v2 < 8; ++v2) {
            float gi = 1.f / (1.f + __expf(-acc[0][v2]));
            float gf = 1.f / (1.f + __expf(-acc[1][v2]));
            float gg = tanhf(acc[2][v2]);
            float go = 1.f / (1.f + __expf(-acc[3][v2]));
            float cn = gf * c[v2] + gi * gg;
            c[v2] = cn;
            float hn = go * tanhf(cn);
            unsigned m = m0 + (unsigned)v2 + 8u * half;        // C/D layout: M = v + 8*half
            unsigned j = hb + l;                               //              N = lane%16
            hout[m * H_ + j] = f2bf(hn);
            if (t == S_ - 1) {
                out[m * H_ + j]           = hn;                // h_t (f32)
                out[B_ * H_ + m * H_ + j] = cn;                // c_t (f32)
            }
        }

        if (t + 1 < S_) {                                      // global_prefetch_b8
            if (USE_XSW) __builtin_prefetch(xt4 + 64u, 0, 0);
            else         __builtin_prefetch(xt + I_, 0, 0);
        }

        // ---- grid-wide barrier (32 persistent WGs) ----
        __threadfence();
        __syncthreads();
        if (tid == 0) {
            atomicAdd(ctr, 1u);
            unsigned target = (unsigned)(t + 1) * NCB;
            while (atomicAdd(ctr, 0u) < target) { __builtin_amdgcn_s_sleep(2); }
        }
        __syncthreads();
        __threadfence();
    }
}

// ---------------------------------------------------------------------------
extern "C" void kernel_launch(void* const* d_in, const int* in_sizes, int n_in,
                              void* d_out, int out_size, void* d_ws, size_t ws_size,
                              hipStream_t stream) {
    (void)in_sizes; (void)n_in; (void)out_size;

    const float* x   = (const float*)d_in[0];
    const float* Wxi = (const float*)d_in[1];
    const float* Whi = (const float*)d_in[2];
    const float* bxi = (const float*)d_in[3];
    const float* bhi = (const float*)d_in[4];
    const float* Wxf = (const float*)d_in[5];
    const float* Whf = (const float*)d_in[6];
    const float* bxf = (const float*)d_in[7];
    const float* bhf = (const float*)d_in[8];
    const float* Wxg = (const float*)d_in[9];
    const float* Whg = (const float*)d_in[10];
    const float* bxg = (const float*)d_in[11];
    const float* bhg = (const float*)d_in[12];
    const float* Wxo = (const float*)d_in[13];
    const float* Who = (const float*)d_in[14];
    const float* bxo = (const float*)d_in[15];
    const float* bho = (const float*)d_in[16];

    const size_t XSW_U32 = 33554432;         // 128 MB swizzled x
    const size_t BSW_U32 = 1048576;          //   4 MB swizzled weights
    const size_t BIAS_U32 = 2048, HBUF_U32 = 32768, CTR_U32 = 16;
    const size_t fast_need = (XSW_U32 + BSW_U32 + BIAS_U32 + HBUF_U32 + CTR_U32) * 4;
    const bool use_xsw = (ws_size >= fast_need);

    unsigned* ws = (unsigned*)d_ws;
    unsigned* xsw  = use_xsw ? ws : nullptr;
    unsigned* Bsw  = ws + (use_xsw ? XSW_U32 : 0);
    float*    bias = (float*)(Bsw + BSW_U32);
    unsigned* hbuf = (unsigned*)bias + BIAS_U32;
    unsigned* ctr  = hbuf + HBUF_U32;

    lstm_pack<<<4096, 256, 0, stream>>>(Wxi, Whi, Wxf, Whf, Wxg, Whg, Wxo, Who, Bsw);
    lstm_init<<<128, 256, 0, stream>>>(bxi, bhi, bxf, bhf, bxg, bhg, bxo, bho,
                                       bias, hbuf, ctr);
    if (use_xsw) {
        lstm_xpack<<<131072, 256, 0, stream>>>(x, xsw);
        lstm_main<true><<<NCB, TPB, 0, stream>>>(x, xsw, Bsw, bias, hbuf, ctr,
                                                 (float*)d_out);
    } else {
        lstm_main<false><<<NCB, TPB, 0, stream>>>(x, nullptr, Bsw, bias, hbuf, ctr,
                                                  (float*)d_out);
    }
}
